// SPVCNN_68152541053541
// MI455X (gfx1250) — compile-verified, matrix operand order checked by reading, not run on hardware
//
// SPVCNN forward for MI455X (gfx1250) — implicit-GEMM WMMA implementation, v3.
//
// Roofline: ~300 GFLOP of conv GEMM (decoder-48^3 dominated) + ~11 GFLOP of
// point linears; activations fit in 192MB L2 -> compute bound -> all matmul
// work on v_wmma_f32_16x16x32_bf16 (bf16 in, fp32 accumulate).
//
// v3: 2Mx2N register blocking (4 accumulators/wave, ~3 issued instr per wmma);
// A-validity handled by a once-per-tap pointer select into a zeroed row (no
// per-chunk cndmasks); transpose-conv reformulated as 8 loop-uniform taps with
// per-row parity zeroing (fixes the B-fragment tap inconsistency) plus a
// wave-uniform __any() skip for inactive taps; B fragments pre-packed per-lane
// (one 32B load); stem input zero-padded 4->32 channels.  BN folded to
// scale/shift and fused into the epilogue with residual / ReLU / mask.

#include <hip/hip_runtime.h>

typedef __bf16 bf16_t;
typedef __attribute__((ext_vector_type(16))) __bf16 v16bf;
typedef __attribute__((ext_vector_type(8)))  float  v8f;

#define WPB 8            // waves per block (wave32)
#define TPB (WPB * 32)

// ---------------------------------------------------------------- utilities
__global__ void k_bnfold(const float* g, const float* b, const float* m, const float* v,
                         float* sc, float* sh, int c) {
    int i = blockIdx.x * blockDim.x + threadIdx.x;
    if (i < c) {
        float s = g[i] * rsqrtf(v[i] + 1e-5f);
        sc[i] = s;
        sh[i] = b[i] - m[i] * s;
    }
}

// pack fp32 weights [taps*CinW, Cout] (jax DHWIO flat) into bf16 WMMA B
// fragments: frag[((tap*KC + ch)*NT + nt)][lane][e], zero-padded in K and N.
__global__ void k_wpack(const float* __restrict__ w, bf16_t* __restrict__ p,
                        int taps, int CinW, int Cout, int KC, int NT, long total) {
    long i = (long)blockIdx.x * blockDim.x + threadIdx.x;
    if (i >= total) return;
    int e = (int)(i & 15);
    long t = i >> 4;
    int lane = (int)(t & 31); t >>= 5;
    int nt = (int)(t % NT); t /= NT;
    int ch = (int)(t % KC); int tap = (int)(t / KC);
    int hi = lane >> 4, ln = lane & 15;
    int k   = ch * 32 + e + ((e < 8) ? 0 : 8) + (hi ? 8 : 0);
    int col = nt * 16 + ln;
    float v = (k < CinW && col < Cout) ? w[((long)tap * CinW + k) * Cout + col] : 0.0f;
    p[i] = (bf16_t)v;
}

// scatter-mean accumulation (fp32 atomics), src either f32 or bf16
__global__ void k_scat_f32(const float* __restrict__ pts, const float* __restrict__ f,
                           float* sum, float* cnt, int gs, float invs, int C, long N) {
    long i = (long)blockIdx.x * blockDim.x + threadIdx.x;
    if (i >= N * (long)C) return;
    long n = i / C; int c = (int)(i - n * C);
    int vx = (int)(pts[n*3+0] * invs); vx = vx < 0 ? 0 : (vx > gs-1 ? gs-1 : vx);
    int vy = (int)(pts[n*3+1] * invs); vy = vy < 0 ? 0 : (vy > gs-1 ? gs-1 : vy);
    int vz = (int)(pts[n*3+2] * invs); vz = vz < 0 ? 0 : (vz > gs-1 ? gs-1 : vz);
    long flat = ((long)vx * gs + vy) * gs + vz;
    atomicAdd(&sum[flat * C + c], f[n * C + c]);
    if (c == 0) atomicAdd(&cnt[flat], 1.0f);
}

__global__ void k_scat_bf(const float* __restrict__ pts, const bf16_t* __restrict__ f,
                          float* sum, float* cnt, int gs, float invs, int C, long N) {
    long i = (long)blockIdx.x * blockDim.x + threadIdx.x;
    if (i >= N * (long)C) return;
    long n = i / C; int c = (int)(i - n * C);
    int vx = (int)(pts[n*3+0] * invs); vx = vx < 0 ? 0 : (vx > gs-1 ? gs-1 : vx);
    int vy = (int)(pts[n*3+1] * invs); vy = vy < 0 ? 0 : (vy > gs-1 ? gs-1 : vy);
    int vz = (int)(pts[n*3+2] * invs); vz = vz < 0 ? 0 : (vz > gs-1 ? gs-1 : vz);
    long flat = ((long)vx * gs + vy) * gs + vz;
    atomicAdd(&sum[flat * C + c], (float)f[n * C + c]);
    if (c == 0) atomicAdd(&cnt[flat], 1.0f);
}

// normalize sums -> bf16 grid (write stride Cpad >= C); optional mask in/out
__global__ void k_norm(const float* __restrict__ sum, const float* __restrict__ cnt,
                       const unsigned char* mIn, bf16_t* out, unsigned char* mOut,
                       long V, int C, int Cpad) {
    long i = (long)blockIdx.x * blockDim.x + threadIdx.x;
    if (i >= V * (long)C) return;
    long v = i / C; int c = (int)(i - v * (long)C);
    float n = cnt[v];
    float val = (n > 0.0f) ? sum[i] / n : 0.0f;
    if (mIn && !mIn[v]) val = 0.0f;
    out[v * (long)Cpad + c] = (bf16_t)val;
    if (mOut && c == 0) mOut[v] = (n > 0.0f) ? 1 : 0;
}

// occupancy pooling: out[D^3] = OR of 2x2x2 children in (2D)^3
__global__ void k_pool(const unsigned char* __restrict__ in, unsigned char* out, int D) {
    long i = (long)blockIdx.x * blockDim.x + threadIdx.x;
    long V = (long)D * D * D;
    if (i >= V) return;
    int z = (int)(i % D); long q = i / D; int y = (int)(q % D); int x = (int)(q / D);
    int D2 = 2 * D;
    unsigned char r = 0;
    #pragma unroll
    for (int j = 0; j < 8; ++j) {
        int ix = 2*x + (j >> 2), iy = 2*y + ((j >> 1) & 1), iz = 2*z + (j & 1);
        r |= in[((long)ix * D2 + iy) * D2 + iz];
    }
    out[i] = r ? 1 : 0;
}

// channel concat [A | B] -> out, channels-last
__global__ void k_concat(const bf16_t* __restrict__ A, int C1,
                         const bf16_t* __restrict__ B, int C2,
                         bf16_t* __restrict__ out, long V) {
    int C = C1 + C2;
    long i = (long)blockIdx.x * blockDim.x + threadIdx.x;
    if (i >= V * (long)C) return;
    long v = i / C; int c = (int)(i - v * (long)C);
    out[i] = (c < C1) ? A[v * (long)C1 + c] : B[v * (long)C2 + (c - C1)];
}

// trilinear devoxelize + optional point-feature add
__global__ void k_devox(const float* __restrict__ pts, const bf16_t* __restrict__ g,
                        const bf16_t* __restrict__ add, bf16_t* __restrict__ out,
                        int gs, float invs, int C, long N) {
    long n = blockIdx.x;
    if (n >= N) return;
    float px = pts[n*3+0] * invs, py = pts[n*3+1] * invs, pz = pts[n*3+2] * invs;
    float fx = floorf(px), fy = floorf(py), fz = floorf(pz);
    int bx = (int)fx, by = (int)fy, bz = (int)fz;
    float rx = px - fx, ry = py - fy, rz = pz - fz;
    for (int c = threadIdx.x; c < C; c += blockDim.x) {
        float acc = add ? (float)add[n * (long)C + c] : 0.0f;
        #pragma unroll
        for (int i = 0; i < 8; ++i) {
            int dx = i >> 2, dy = (i >> 1) & 1, dz = i & 1;
            int ix = bx + dx, iy = by + dy, iz = bz + dz;
            bool v = (unsigned)ix < (unsigned)gs && (unsigned)iy < (unsigned)gs &&
                     (unsigned)iz < (unsigned)gs;
            float w = (dx ? rx : 1.0f - rx) * (dy ? ry : 1.0f - ry) * (dz ? rz : 1.0f - rz);
            if (v) acc += w * (float)g[(((long)ix * gs + iy) * gs + iz) * C + c];
        }
        out[n * (long)C + c] = (bf16_t)acc;
    }
}

// ------------------------------------------- implicit-GEMM conv on WMMA bf16
// mode: 0 = 3x3x3 SAME, 1 = 1x1x1, 2 = ks2/s2 down, 3 = ks2/s2 transpose up
// Each wave: 32 M rows (2 tiles) x 32 N cols (2 tiles), 4 accumulators.
struct ConvArgs {
    const bf16_t* in;            // [Din^3, Cin] bf16, Cin % 32 == 0, 16B rows
    const bf16_t* w;             // packed B fragments (k_wpack layout)
    const float*  sc;            // folded BN scale [Cout] (nullable)
    const float*  sh;            // folded BN shift [Cout]
    const bf16_t* resid;         // residual [Dout^3, Cout] (nullable; may alias out)
    const unsigned char* mask;   // occupancy [Dout^3] (nullable)
    const bf16_t* zrow;          // zeroed scratch row (>= Cin+32 elems)
    bf16_t* out;                 // [Dout^3, Cout]
    int Din, Dout, Cin, Cout, mode, relu, KC, NT;
};

__global__ __launch_bounds__(TPB) void k_conv_wmma(ConvArgs a) {
    const int lane = threadIdx.x & 31;
    const int wv   = threadIdx.x >> 5;
    const long M   = (long)a.Dout * a.Dout * a.Dout;
    const long mBase = ((long)blockIdx.x * WPB + wv) * 32;
    if (mBase >= M) return;                       // wave-uniform exit
    const int ln15 = lane & 15;                   // A row index / B col index
    const int hi   = lane >> 4;
    long m0 = mBase + ln15, m1 = m0 + 16;
    const bool v0m = (m0 < M), v1m = (m1 < M);
    long c0m = v0m ? m0 : 0, c1m = v1m ? m1 : 0;
    const int D = a.Dout;
    int z0 = (int)(c0m % D); long q0 = c0m / D; int y0 = (int)(q0 % D); int x0 = (int)(q0 / D);
    int z1 = (int)(c1m % D); long q1 = c1m / D; int y1 = (int)(q1 % D); int x1 = (int)(q1 / D);

    v8f a00, a01, a10, a11;
    #pragma unroll
    for (int r = 0; r < 8; ++r) { a00[r] = 0.f; a01[r] = 0.f; a10[r] = 0.f; a11[r] = 0.f; }

    const int Din = a.Din, KC = a.KC, NT = a.NT, Cin = a.Cin;
    const long chStride = (long)NT * 512;          // packed elems per K-chunk step
    const bf16_t* wl = a.w + (long)blockIdx.y * 1024 + lane * 16;  // nt0 = 2*blockIdx.y
    const bf16_t* zr = a.zrow + hi * 8;

    const int ntap = (a.mode == 0) ? 27 : ((a.mode == 1) ? 1 : 8);
    for (int t = 0; t < ntap; ++t) {
        long n0i = 0, n1i = 0; bool nv0 = v0m, nv1 = v1m;
        if (a.mode == 0) {
            int dx = t / 9 - 1, dy = (t / 3) % 3 - 1, dz = t % 3 - 1;
            int ix0 = x0 + dx, iy0 = y0 + dy, iz0 = z0 + dz;
            nv0 = nv0 && (unsigned)ix0 < (unsigned)Din && (unsigned)iy0 < (unsigned)Din &&
                         (unsigned)iz0 < (unsigned)Din;
            n0i = nv0 ? ((long)ix0 * Din + iy0) * Din + iz0 : 0;
            int ix1 = x1 + dx, iy1 = y1 + dy, iz1 = z1 + dz;
            nv1 = nv1 && (unsigned)ix1 < (unsigned)Din && (unsigned)iy1 < (unsigned)Din &&
                         (unsigned)iz1 < (unsigned)Din;
            n1i = nv1 ? ((long)ix1 * Din + iy1) * Din + iz1 : 0;
        } else if (a.mode == 1) {
            n0i = ((long)x0 * Din + y0) * Din + z0;
            n1i = ((long)x1 * Din + y1) * Din + z1;
        } else if (a.mode == 2) {
            int tx = t >> 2, ty = (t >> 1) & 1, tz = t & 1;
            n0i = ((long)(2*x0 + tx) * Din + (2*y0 + ty)) * Din + (2*z0 + tz);
            n1i = ((long)(2*x1 + tx) * Din + (2*y1 + ty)) * Din + (2*z1 + tz);
        } else { // transpose up: tap is loop-uniform; rows gated by parity match
            int tx = t >> 2, ty = (t >> 1) & 1, tz = t & 1;
            nv0 = nv0 && ((x0 & 1) == 1 - tx) && ((y0 & 1) == 1 - ty) && ((z0 & 1) == 1 - tz);
            nv1 = nv1 && ((x1 & 1) == 1 - tx) && ((y1 & 1) == 1 - ty) && ((z1 & 1) == 1 - tz);
            n0i = ((long)(x0 >> 1) * Din + (y0 >> 1)) * Din + (z0 >> 1);
            n1i = ((long)(x1 >> 1) * Din + (y1 >> 1)) * Din + (z1 >> 1);
        }
        if (!__any(nv0 || nv1)) continue;          // wave-uniform tap skip
        // validity folded into a pointer select (zeroed row) -> clean K loop
        const bf16_t* r0 = nv0 ? (a.in + n0i * (long)Cin + hi * 8) : zr;
        const bf16_t* r1 = nv1 ? (a.in + n1i * (long)Cin + hi * 8) : zr;
        const bf16_t* f0 = wl + (long)t * KC * chStride;
        const bf16_t* f1 = f0 + 512;
        __builtin_prefetch(f0, 0, 1);
        for (int ch = 0; ch < KC; ++ch) {
            union { v16bf v; uint4 q[2]; } A0, A1;
            A0.q[0] = *(const uint4*)(r0); A0.q[1] = *(const uint4*)(r0 + 16);
            A1.q[0] = *(const uint4*)(r1); A1.q[1] = *(const uint4*)(r1 + 16);
            v16bf B0 = *(const v16bf*)(f0);
            v16bf B1 = *(const v16bf*)(f1);
            a00 = __builtin_amdgcn_wmma_f32_16x16x32_bf16(false, A0.v, false, B0,
                                                          (short)0, a00, false, false);
            a01 = __builtin_amdgcn_wmma_f32_16x16x32_bf16(false, A0.v, false, B1,
                                                          (short)0, a01, false, false);
            a10 = __builtin_amdgcn_wmma_f32_16x16x32_bf16(false, A1.v, false, B0,
                                                          (short)0, a10, false, false);
            a11 = __builtin_amdgcn_wmma_f32_16x16x32_bf16(false, A1.v, false, B1,
                                                          (short)0, a11, false, false);
            r0 += 32; r1 += 32; f0 += chStride; f1 += chStride;
        }
    }

    // epilogue: BN scale/shift -> residual -> ReLU -> occupancy mask -> bf16
    const int nb = blockIdx.y * 32 + ln15;
    auto epi = [&](v8f& ac, long mt, int n) {
        if (n >= a.Cout) return;
        float scl = a.sc ? a.sc[n] : 1.0f;
        float shf = a.sc ? a.sh[n] : 0.0f;
        #pragma unroll
        for (int r = 0; r < 8; ++r) {
            long mm = mt + r + (hi ? 8 : 0);
            if (mm < M) {
                float val = ac[r] * scl + shf;
                if (a.resid) val += (float)a.resid[mm * (long)a.Cout + n];
                if (a.relu)  val = fmaxf(val, 0.0f);
                if (a.mask && !a.mask[mm]) val = 0.0f;
                a.out[mm * (long)a.Cout + n] = (bf16_t)val;
            }
        }
    };
    epi(a00, mBase,      nb);
    epi(a01, mBase,      nb + 16);
    epi(a10, mBase + 16, nb);
    epi(a11, mBase + 16, nb + 16);
}

// --------------------------------------------------- point linear on WMMA
struct LinArgs {
    const bf16_t* x;      // [N, Cin] bf16, Cin % 32 == 0
    const bf16_t* w;      // packed B fragments (taps = 1)
    const float*  bias;   // [Cout] (nullable)
    const float*  sc;     // folded BN (nullable)
    const float*  sh;
    const bf16_t* zrow;   // zeroed scratch row
    bf16_t* out;          // bf16 output (nullable)
    float*  outf;         // fp32 output (nullable) — used for the classifier
    long N; int Cin, Cout, relu, KC, NT;
};

__global__ __launch_bounds__(TPB) void k_lin_wmma(LinArgs a) {
    const int lane = threadIdx.x & 31;
    const int wv   = threadIdx.x >> 5;
    const long mBase = ((long)blockIdx.x * WPB + wv) * 32;
    if (mBase >= a.N) return;
    const int ln15 = lane & 15;
    const int hi   = lane >> 4;
    long m0 = mBase + ln15, m1 = m0 + 16;
    const bool v0m = (m0 < a.N), v1m = (m1 < a.N);

    v8f a00, a01, a10, a11;
    #pragma unroll
    for (int r = 0; r < 8; ++r) { a00[r] = 0.f; a01[r] = 0.f; a10[r] = 0.f; a11[r] = 0.f; }

    const long chStride = (long)a.NT * 512;
    const bf16_t* zr = a.zrow + hi * 8;
    const bf16_t* r0 = v0m ? (a.x + m0 * (long)a.Cin + hi * 8) : zr;
    const bf16_t* r1 = v1m ? (a.x + m1 * (long)a.Cin + hi * 8) : zr;
    const bf16_t* f0 = a.w + (long)blockIdx.y * 1024 + lane * 16;
    const bf16_t* f1 = f0 + 512;
    __builtin_prefetch(f0, 0, 1);
    for (int ch = 0; ch < a.KC; ++ch) {
        union { v16bf v; uint4 q[2]; } A0, A1;
        A0.q[0] = *(const uint4*)(r0); A0.q[1] = *(const uint4*)(r0 + 16);
        A1.q[0] = *(const uint4*)(r1); A1.q[1] = *(const uint4*)(r1 + 16);
        v16bf B0 = *(const v16bf*)(f0);
        v16bf B1 = *(const v16bf*)(f1);
        a00 = __builtin_amdgcn_wmma_f32_16x16x32_bf16(false, A0.v, false, B0,
                                                      (short)0, a00, false, false);
        a01 = __builtin_amdgcn_wmma_f32_16x16x32_bf16(false, A0.v, false, B1,
                                                      (short)0, a01, false, false);
        a10 = __builtin_amdgcn_wmma_f32_16x16x32_bf16(false, A1.v, false, B0,
                                                      (short)0, a10, false, false);
        a11 = __builtin_amdgcn_wmma_f32_16x16x32_bf16(false, A1.v, false, B1,
                                                      (short)0, a11, false, false);
        r0 += 32; r1 += 32; f0 += chStride; f1 += chStride;
    }

    const int nb = blockIdx.y * 32 + ln15;
    auto epi = [&](v8f& ac, long mt, int n) {
        if (n >= a.Cout) return;
        float bs  = a.bias ? a.bias[n] : 0.0f;
        float scl = a.sc ? a.sc[n] : 1.0f;
        float shf = a.sc ? a.sh[n] : 0.0f;
        #pragma unroll
        for (int r = 0; r < 8; ++r) {
            long mm = mt + r + (hi ? 8 : 0);
            if (mm < a.N) {
                float val = (ac[r] + bs) * scl + shf;   // x@w + b, then BN
                if (a.relu) val = fmaxf(val, 0.0f);
                if (a.outf) a.outf[mm * (long)a.Cout + n] = val;
                else        a.out [mm * (long)a.Cout + n] = (bf16_t)val;
            }
        }
    };
    epi(a00, mBase,      nb);
    epi(a01, mBase,      nb + 16);
    epi(a10, mBase + 16, nb);
    epi(a11, mBase + 16, nb + 16);
}

// ================================================================== host
struct BNh  { const float *b, *g, *m, *v; int c; };
struct ConvW { const float* w; BNh bn; int taps, cin, cout; };
struct ResW  { ConvW c1, c2, cd; bool wd; };
struct LinW  { const float* b; BNh bn; const float* w; int cin, cout; };
struct DevConv { const bf16_t* w; const float* sc; const float* sh;
                 int taps, cin, cout, KC, NT; };
struct DevRes  { DevConv c1, c2, cd; bool wd; };
struct DevLin  { const bf16_t* w; const float* bias; const float* sc; const float* sh;
                 int cin, cout, KC, NT; };

extern "C" void kernel_launch(void* const* d_in, const int* in_sizes, int n_in,
                              void* d_out, int out_size, void* d_ws, size_t ws_size,
                              hipStream_t stream) {
    (void)in_sizes; (void)n_in; (void)out_size;
    if (ws_size < ((size_t)384 << 20)) return;   // need ~350 MB of scratch

    const long NP = 100000;
    const int  CS[9] = {32, 32, 64, 128, 256, 256, 128, 96, 96};
    const long V0 = 48L*48*48, V1 = 24L*24*24, V2 = 12L*12*12, V3 = 6L*6*6, V4 = 27;

    // ---- parameter walk: d_in[0]=points, d_in[1]=feats, then jax sorted-key
    // leaves: cls, down0..3, pt0..2, stem0..1, up0..3 (bn dicts -> b,g,m,v).
    int ip = 0;
    const float* pts   = (const float*)d_in[ip++];
    const float* feats = (const float*)d_in[ip++];
    auto nx = [&]() { return (const float*)d_in[ip++]; };
    auto getBN = [&](int c) { BNh p; p.b = nx(); p.g = nx(); p.m = nx(); p.v = nx();
                              p.c = c; return p; };
    auto getRES = [&](int cin, int cout) {
        ResW r; r.wd = (cin != cout);
        r.c1.bn = getBN(cout); r.c2.bn = getBN(cout);
        if (r.wd) r.cd.bn = getBN(cout);
        r.c1.w = nx(); r.c2.w = nx();
        if (r.wd) r.cd.w = nx(); else r.cd.w = nullptr;
        r.c1.taps = 27; r.c1.cin = cin;  r.c1.cout = cout;
        r.c2.taps = 27; r.c2.cin = cout; r.c2.cout = cout;
        r.cd.taps = 1;  r.cd.cin = cin;  r.cd.cout = cout;
        return r;
    };
    auto getLIN = [&](int cin, int cout) { LinW l; l.b = nx(); l.bn = getBN(cout);
                                           l.w = nx(); l.cin = cin; l.cout = cout;
                                           return l; };

    const float* cls_b = nx();
    const float* cls_w = nx();
    ConvW dn_c[4]; ResW dn_r1[4], dn_r2[4];
    for (int i = 0; i < 4; ++i) {
        dn_c[i].bn = getBN(CS[i]); dn_c[i].w = nx();
        dn_c[i].taps = 8; dn_c[i].cin = CS[i]; dn_c[i].cout = CS[i];
        dn_r1[i] = getRES(CS[i], CS[i+1]);
        dn_r2[i] = getRES(CS[i+1], CS[i+1]);
    }
    LinW pt0 = getLIN(CS[0], CS[4]);
    LinW pt1 = getLIN(CS[4], CS[6]);
    LinW pt2 = getLIN(CS[6], CS[8]);
    ConvW st0; st0.bn = getBN(CS[0]); st0.w = nx(); st0.taps = 27; st0.cin = 4;  st0.cout = CS[0];
    ConvW st1; st1.bn = getBN(CS[0]); st1.w = nx(); st1.taps = 27; st1.cin = CS[0]; st1.cout = CS[0];
    ConvW up_d[4]; ResW up_r1[4], up_r2[4];
    for (int j = 0; j < 4; ++j) {
        int i = 4 + j;
        up_d[j].bn = getBN(CS[i+1]); up_d[j].w = nx();
        up_d[j].taps = 8; up_d[j].cin = CS[i]; up_d[j].cout = CS[i+1];
        up_r1[j] = getRES(CS[i+1] + CS[7-i], CS[i+1]);
        up_r2[j] = getRES(CS[i+1], CS[i+1]);
    }

    // ---- workspace bump allocator
    char* wsp = (char*)d_ws; size_t off = 0;
    auto alloc = [&](size_t bytes) -> void* {
        void* p = wsp + off; off = (off + bytes + 255) & ~(size_t)255; return p;
    };

    // ---- device-side layer prep: pack weights into WMMA fragments, fold BN
    auto packW = [&](const float* w, int taps, int cinW, int cout,
                     int& KC, int& NT) -> bf16_t* {
        int cinP = (cinW + 31) & ~31;
        KC = cinP / 32; NT = (cout + 15) / 16;
        long total = (long)taps * KC * NT * 512;
        bf16_t* wb = (bf16_t*)alloc(total * sizeof(bf16_t));
        k_wpack<<<dim3((unsigned)((total + 255) / 256)), dim3(256), 0, stream>>>(
            w, wb, taps, cinW, cout, KC, NT, total);
        return wb;
    };
    auto foldBN = [&](const BNh& bn, const float*& sc, const float*& sh) {
        float* s = (float*)alloc(bn.c * 4);
        float* h = (float*)alloc(bn.c * 4);
        k_bnfold<<<dim3((unsigned)((bn.c + 63) / 64)), dim3(64), 0, stream>>>(
            bn.g, bn.b, bn.m, bn.v, s, h, bn.c);
        sc = s; sh = h;
    };
    auto prep = [&](const ConvW& c) -> DevConv {
        DevConv d;
        d.w = packW(c.w, c.taps, c.cin, c.cout, d.KC, d.NT);
        foldBN(c.bn, d.sc, d.sh);
        d.taps = c.taps; d.cin = (c.cin + 31) & ~31; d.cout = c.cout;
        return d;
    };
    auto prepRes = [&](const ResW& r) -> DevRes {
        DevRes d; d.wd = r.wd; d.c1 = prep(r.c1); d.c2 = prep(r.c2);
        if (r.wd) d.cd = prep(r.cd);
        else d.cd = DevConv{nullptr, nullptr, nullptr, 0, 0, 0, 0, 0};
        return d;
    };
    auto prepLin = [&](const LinW& l) -> DevLin {
        DevLin d;
        d.w = packW(l.w, 1, l.cin, l.cout, d.KC, d.NT);
        foldBN(l.bn, d.sc, d.sh);
        d.bias = l.b; d.cin = l.cin; d.cout = l.cout;
        return d;
    };

    DevConv d_st0 = prep(st0), d_st1 = prep(st1);
    DevConv d_dn_c[4]; DevRes d_dn_r1[4], d_dn_r2[4];
    for (int i = 0; i < 4; ++i) { d_dn_c[i] = prep(dn_c[i]);
                                  d_dn_r1[i] = prepRes(dn_r1[i]);
                                  d_dn_r2[i] = prepRes(dn_r2[i]); }
    DevConv d_up_d[4]; DevRes d_up_r1[4], d_up_r2[4];
    for (int j = 0; j < 4; ++j) { d_up_d[j] = prep(up_d[j]);
                                  d_up_r1[j] = prepRes(up_r1[j]);
                                  d_up_r2[j] = prepRes(up_r2[j]); }
    DevLin d_pt0 = prepLin(pt0), d_pt1 = prepLin(pt1), d_pt2 = prepLin(pt2);
    DevLin d_cls;
    d_cls.w = packW(cls_w, 1, CS[8], 19, d_cls.KC, d_cls.NT);
    d_cls.bias = cls_b; d_cls.sc = nullptr; d_cls.sh = nullptr;
    d_cls.cin = CS[8]; d_cls.cout = 19;

    // ---- activation buffers (bf16) + scatter scratch (fp32) + masks (u8)
    bf16_t* ZROW = (bf16_t*)alloc(2048);            // zeroed A row for invalid taps
    bf16_t* T0 = (bf16_t*)alloc(V0 * 32  * sizeof(bf16_t));
    bf16_t* T1 = (bf16_t*)alloc(V0 * 128 * sizeof(bf16_t));
    bf16_t* T2 = (bf16_t*)alloc(V0 * 128 * sizeof(bf16_t));
    bf16_t* T3 = (bf16_t*)alloc(V0 * 128 * sizeof(bf16_t));
    bf16_t* T4 = (bf16_t*)alloc(V0 * 128 * sizeof(bf16_t));
    bf16_t* X0 = (bf16_t*)alloc(V0 * 32  * sizeof(bf16_t));
    bf16_t* FV1 = (bf16_t*)alloc(V1 * 32  * sizeof(bf16_t));
    bf16_t* FV2 = (bf16_t*)alloc(V2 * 64  * sizeof(bf16_t));
    bf16_t* FV3 = (bf16_t*)alloc(V3 * 128 * sizeof(bf16_t));
    bf16_t* FV4 = (bf16_t*)alloc(V4 * 256 * sizeof(bf16_t));
    bf16_t* Z0 = (bf16_t*)alloc(NP * 32  * sizeof(bf16_t));
    bf16_t* Z1 = (bf16_t*)alloc(NP * 256 * sizeof(bf16_t));
    bf16_t* Z2 = (bf16_t*)alloc(NP * 128 * sizeof(bf16_t));
    bf16_t* Z3 = (bf16_t*)alloc(NP * 96  * sizeof(bf16_t));
    bf16_t* PT = (bf16_t*)alloc(NP * 256 * sizeof(bf16_t));
    float* SSUM = (float*)alloc(V0 * 32 * 4);
    float* CNT  = (float*)alloc(V0 * 4);
    unsigned char* M0 = (unsigned char*)alloc(V0);
    unsigned char* M1 = (unsigned char*)alloc(V1);
    unsigned char* M2 = (unsigned char*)alloc(V2);
    unsigned char* M3 = (unsigned char*)alloc(V3);
    unsigned char* M4 = (unsigned char*)alloc(V4);
    hipMemsetAsync(ZROW, 0, 2048, stream);

    // ---- launch helpers
    auto conv = [&](const DevConv& d, const bf16_t* in, int Din, int Dout, int mode,
                    const bf16_t* resid, const unsigned char* mask, int relu, bf16_t* out) {
        ConvArgs a; a.in = in; a.w = d.w; a.sc = d.sc; a.sh = d.sh; a.resid = resid;
        a.mask = mask; a.zrow = ZROW; a.out = out; a.Din = Din; a.Dout = Dout;
        a.Cin = d.cin; a.Cout = d.cout; a.mode = mode; a.relu = relu;
        a.KC = d.KC; a.NT = d.NT;
        long M = (long)Dout * Dout * Dout;
        dim3 gr((unsigned)((M + 32 * WPB - 1) / (32 * WPB)), (unsigned)(d.NT / 2));
        k_conv_wmma<<<gr, dim3(TPB), 0, stream>>>(a);
    };
    // ResidualBlock: conv-bn-relu-mask -> conv-bn (+sc) -> relu -> mask
    auto res = [&](const DevRes& R, const bf16_t* x, int D, const unsigned char* mask,
                   bf16_t* t1, bf16_t* tsc, bf16_t* out) {
        conv(R.c1, x, D, D, 0, nullptr, mask, 1, t1);
        const bf16_t* sc = x;
        if (R.wd) { conv(R.cd, x, D, D, 1, nullptr, nullptr, 0, tsc); sc = tsc; }
        conv(R.c2, t1, D, D, 0, sc, mask, 1, out);   // in-place resid (out==tsc) is safe
    };
    auto lin = [&](const DevLin& L, const bf16_t* x, bf16_t* out, float* outf, int relu) {
        LinArgs a; a.x = x; a.w = L.w; a.bias = L.bias; a.sc = L.sc; a.sh = L.sh;
        a.zrow = ZROW; a.out = out; a.outf = outf; a.N = NP; a.Cin = L.cin;
        a.Cout = L.cout; a.relu = relu; a.KC = L.KC; a.NT = L.NT;
        dim3 gr((unsigned)((NP + 32 * WPB - 1) / (32 * WPB)), (unsigned)(L.NT / 2));
        k_lin_wmma<<<gr, dim3(TPB), 0, stream>>>(a);
    };
    auto scatter = [&](const void* feat, bool f32src, int gs, float stride, int C,
                       int Cpad, const unsigned char* mIn, bf16_t* outGrid,
                       unsigned char* mOut) {
        long V = (long)gs * gs * gs;
        hipMemsetAsync(SSUM, 0, (size_t)V * C * 4, stream);
        hipMemsetAsync(CNT, 0, (size_t)V * 4, stream);
        long tot = NP * (long)C;
        dim3 gr((unsigned)((tot + 255) / 256));
        if (f32src)
            k_scat_f32<<<gr, dim3(256), 0, stream>>>(pts, (const float*)feat, SSUM, CNT,
                                                     gs, 1.0f / stride, C, NP);
        else
            k_scat_bf<<<gr, dim3(256), 0, stream>>>(pts, (const bf16_t*)feat, SSUM, CNT,
                                                    gs, 1.0f / stride, C, NP);
        long vt = V * (long)C;
        k_norm<<<dim3((unsigned)((vt + 255) / 256)), dim3(256), 0, stream>>>(
            SSUM, CNT, mIn, outGrid, mOut, V, C, Cpad);
    };
    auto devox = [&](const bf16_t* g, const bf16_t* add, bf16_t* out, int gs,
                     float stride, int C) {
        k_devox<<<dim3((unsigned)NP), dim3(128), 0, stream>>>(pts, g, add, out, gs,
                                                              1.0f / stride, C, NP);
    };

    // ================================================== forward schedule
    // initial voxelize: C=4 zero-padded into a 32-channel grid -> mask0 ; stem
    hipMemsetAsync(T1, 0, (size_t)V0 * 32 * sizeof(bf16_t), stream);
    scatter(feats, true, 48, 1.0f, 4, 32, nullptr, T1, M0);
    conv(d_st0, T1, 48, 48, 0, nullptr, M0, 1, T0);
    conv(d_st1, T0, 48, 48, 0, nullptr, M0, 1, X0);
    devox(X0, nullptr, Z0, 48, 1.0f, 32);
    scatter(Z0, false, 48, 1.0f, 32, 32, M0, T0, nullptr);  // fv[0] = vgrid in T0

    unsigned char* Ms[5] = {M0, M1, M2, M3, M4};
    int Ds[5] = {48, 24, 12, 6, 3};
    bf16_t* FVs[5] = {T0, FV1, FV2, FV3, FV4};
    for (int i = 0; i < 4; ++i) {
        long Vn = (long)Ds[i+1] * Ds[i+1] * Ds[i+1];
        k_pool<<<dim3((unsigned)((Vn + 255) / 256)), dim3(256), 0, stream>>>(
            Ms[i], Ms[i+1], Ds[i+1]);
        conv(d_dn_c[i], FVs[i], Ds[i], Ds[i+1], 2, nullptr, Ms[i+1], 1, T1);
        res(d_dn_r1[i], T1, Ds[i+1], Ms[i+1], T2, T3, T3);
        res(d_dn_r2[i], T3, Ds[i+1], Ms[i+1], T2, T4, FVs[i+1]);
    }

    // z1 = devox(fv4, /16) + pt0(z0) ; re-voxelize at level 4
    lin(d_pt0, Z0, PT, nullptr, 1);
    devox(FV4, PT, Z1, 3, 16.0f, 256);
    scatter(Z1, false, 3, 16.0f, 256, 256, M4, T0, nullptr);

    // up0: 3 -> 6, skip fv3
    conv(d_up_d[0], T0, 3, 6, 3, nullptr, M3, 1, T1);
    k_concat<<<dim3((unsigned)((V3 * 384 + 255) / 256)), dim3(256), 0, stream>>>(
        T1, 256, FV3, 128, T2, V3);
    res(d_up_r1[0], T2, 6, M3, T1, T3, T3);
    res(d_up_r2[0], T3, 6, M3, T1, T4, T4);

    // up1: 6 -> 12, skip fv2
    conv(d_up_d[1], T4, 6, 12, 3, nullptr, M2, 1, T1);
    k_concat<<<dim3((unsigned)((V2 * 192 + 255) / 256)), dim3(256), 0, stream>>>(
        T1, 128, FV2, 64, T2, V2);
    res(d_up_r1[1], T2, 12, M2, T1, T3, T3);
    res(d_up_r2[1], T3, 12, M2, T1, T0, T0);                // y12 in T0

    // z2 = devox(y12, /4) + pt1(z1) ; re-voxelize at level 2
    lin(d_pt1, Z1, PT, nullptr, 1);
    devox(T0, PT, Z2, 12, 4.0f, 128);
    scatter(Z2, false, 12, 4.0f, 128, 128, M2, T1, nullptr);

    // up2: 12 -> 24, skip fv1
    conv(d_up_d[2], T1, 12, 24, 3, nullptr, M1, 1, T2);
    k_concat<<<dim3((unsigned)((V1 * 128 + 255) / 256)), dim3(256), 0, stream>>>(
        T2, 96, FV1, 32, T3, V1);
    res(d_up_r1[2], T3, 24, M1, T2, T4, T4);
    res(d_up_r2[2], T4, 24, M1, T2, T0, T0);                // y24 in T0

    // up3: 24 -> 48, skip x0
    conv(d_up_d[3], T0, 24, 48, 3, nullptr, M0, 1, T1);
    k_concat<<<dim3((unsigned)((V0 * 128 + 255) / 256)), dim3(256), 0, stream>>>(
        T1, 96, X0, 32, T2, V0);
    res(d_up_r1[3], T2, 48, M0, T1, T3, T3);
    res(d_up_r2[3], T3, 48, M0, T1, T4, T4);                // final voxel features

    // z3 = devox(final, /1) + pt2(z2) ; classifier -> fp32 d_out
    lin(d_pt2, Z2, PT, nullptr, 1);
    devox(T4, PT, Z3, 48, 1.0f, 96);
    lin(d_cls, Z3, nullptr, (float*)d_out, 0);
}
